// PatchEmbedHier_37134287241639
// MI455X (gfx1250) — compile-verified, matrix operand order checked by reading, not run on hardware
//
#include <hip/hip_runtime.h>
#include <hip/hip_bf16.h>

typedef __bf16 bf16;
typedef __attribute__((ext_vector_type(16))) __bf16 v16bf;
typedef __attribute__((ext_vector_type(8)))  float  v8f;
typedef int i4vec __attribute__((vector_size(16)));   // int4, matches builtin param

#define Bn  4
#define Np  16384
#define G1n 1024
#define K1n 32
#define G2n 256
#define K2n 16

// CDNA5 async global->LDS copy (ASYNCcnt path), guarded so a missing builtin
// falls back to a plain VGPR-staged copy.
#if defined(__gfx1250__) && __has_builtin(__builtin_amdgcn_global_load_async_to_lds_b128)
#define HAVE_ASYNC_LDS 1
#endif

// ---------------------------------------------------------------------------
// Weight prep: convert f32 -> bf16 AND pre-swizzle into the per-lane WMMA
// B-fragment layout, so the GEMM K-loop loads B as two contiguous b128 per
// lane straight from global (L2-hot), with zero LDS traffic / unpack ALU.
// Packed index: (((nt*(K/32)+kt)*32 + lane)*16 + e), where element e of the
// v16bf fragment is B[k][n] with n = nt*16 + (lane&15) and
// k = kt*32 + ((j>>2)<<4) + ((lane>>4)<<3) + ((j&3)<<1) + (e&1), j = e>>1.
// Rows k >= Ksrc are zero-padding (used for e2_w1: 131 -> 160).
// ---------------------------------------------------------------------------
__global__ void pack_w_kernel(const float* __restrict__ s, bf16* __restrict__ d,
                              int K, int Ksrc, int N) {
    int i = blockIdx.x * 256 + threadIdx.x;
    if (i >= K * N) return;
    int e    = i & 15;
    int l    = (i >> 4) & 31;
    int t    = i >> 9;                   // nt*(K/32) + kt
    int kt   = t % (K / 32);
    int nt   = t / (K / 32);
    int j    = e >> 1;
    int half = l >> 4;
    int k = kt * 32 + ((j >> 2) << 4) + (half << 3) + ((j & 3) << 1) + (e & 1);
    int n = nt * 16 + (l & 15);
    d[i] = (k < Ksrc) ? (bf16)s[k * N + n] : (bf16)0.0f;
}

// ---------------------------------------------------------------------------
// FPS: one block per batch, 1024 threads, coords register-resident (16/thread).
// Sequential chain of 1024 (update min_d -> argmax) steps; packed u64 keys give
// exact first-occurrence tie-break like jnp.argmax.
// ---------------------------------------------------------------------------
__global__ void __launch_bounds__(1024) fps_kernel(const float* __restrict__ coords,
                                                   float* __restrict__ centers1) {
    __shared__ unsigned long long s_bk[32];
    __shared__ int s_cur;
    const int t = threadIdx.x, b = blockIdx.x;
    const int lane = t & 31, wid = t >> 5;
    const float* C = coords + (size_t)b * Np * 3;

    float px[16], py[16], pz[16], md[16];
#pragma unroll
    for (int i = 0; i < 16; ++i) {
        int p = i * 1024 + t;
        px[i] = C[p * 3 + 0]; py[i] = C[p * 3 + 1]; pz[i] = C[p * 3 + 2];
        md[i] = 1.0e10f;
    }
    int cur = 0;
    for (int g = 0; g < G1n; ++g) {
        float cx = C[cur * 3 + 0], cy = C[cur * 3 + 1], cz = C[cur * 3 + 2];
        if (t == 0) {
            float* o = centers1 + (size_t)(b * G1n + g) * 3;
            o[0] = cx; o[1] = cy; o[2] = cz;
        }
        unsigned long long bk = 0;
#pragma unroll
        for (int i = 0; i < 16; ++i) {
            float dx = px[i] - cx, dy = py[i] - cy, dz = pz[i] - cz;
            float d = dx * dx + dy * dy + dz * dz;
            md[i] = fminf(md[i], d);
            unsigned long long key =
                ((unsigned long long)__float_as_uint(md[i]) << 32) |
                (unsigned)(~(unsigned)(i * 1024 + t));
            if (key > bk) bk = key;
        }
#pragma unroll
        for (int off = 16; off > 0; off >>= 1) {
            unsigned long long o = __shfl_xor(bk, off, 32);
            if (o > bk) bk = o;
        }
        if (lane == 0) s_bk[wid] = bk;
        __syncthreads();
        if (t < 32) {
            unsigned long long k = s_bk[t];
#pragma unroll
            for (int off = 16; off > 0; off >>= 1) {
                unsigned long long o = __shfl_xor(k, off, 32);
                if (o > k) k = o;
            }
            if (t == 0) s_cur = (int)(~(unsigned)(k & 0xffffffffu));
        }
        __syncthreads();
        cur = s_cur;
    }
}

// ---------------------------------------------------------------------------
// kNN selection: one wave per query. Each lane keeps a register top-KOUT over
// its strided subset (compile-time-indexed -> cndmask, no scratch), then KOUT
// rounds of wave-wide packed u64 min-merge. Lane r returns the r-th nearest.
// ---------------------------------------------------------------------------
template <int NPTS, int KOUT>
__device__ __forceinline__ int select_knn(const float* __restrict__ P,
                                          float cx, float cy, float cz, int lane) {
    float dtop[KOUT]; int itop[KOUT];
#pragma unroll
    for (int i = 0; i < KOUT; ++i) { dtop[i] = 3.0e38f; itop[i] = 0; }
    float curmax = 3.0e38f; int maxpos = 0;

    for (int j = 0; j < NPTS / 32; ++j) {
        int p = j * 32 + lane;
        float dx = P[p * 3 + 0] - cx, dy = P[p * 3 + 1] - cy, dz = P[p * 3 + 2] - cz;
        float d = dx * dx + dy * dy + dz * dz;
        if (d < curmax) {
#pragma unroll
            for (int i = 0; i < KOUT; ++i) if (i == maxpos) { dtop[i] = d; itop[i] = p; }
            float mv = -1.0f; int mp = 0;
#pragma unroll
            for (int i = 0; i < KOUT; ++i) if (dtop[i] > mv) { mv = dtop[i]; mp = i; }
            curmax = mv; maxpos = mp;
        }
    }
    int mynbr = 0;
#pragma unroll 1
    for (int r = 0; r < KOUT; ++r) {
        float mv = 3.0e38f; int ms = 0; unsigned mi = 0;
#pragma unroll
        for (int i = 0; i < KOUT; ++i)
            if (dtop[i] < mv) { mv = dtop[i]; ms = i; mi = (unsigned)itop[i]; }
        unsigned long long key = ((unsigned long long)__float_as_uint(mv) << 32) | mi;
        unsigned long long mykey = key;
#pragma unroll
        for (int off = 16; off > 0; off >>= 1) {
            unsigned long long o = __shfl_xor(key, off, 32);
            if (o < key) key = o;
        }
        if (key == mykey) {
#pragma unroll
            for (int i = 0; i < KOUT; ++i) if (i == ms) dtop[i] = 3.0e38f;
        }
        if (lane == r) mynbr = (int)(unsigned)(key & 0xffffffffu);
    }
    return mynbr;
}

// kNN-1 + gather + fused MLP layer1 (6->64, relu) -> h1 bf16 (B*G1*K1, 64)
__global__ void __launch_bounds__(256) knn1_kernel(const float* __restrict__ coords,
                                                   const float* __restrict__ feats,
                                                   const float* __restrict__ centers1,
                                                   const float* __restrict__ w1,
                                                   const float* __restrict__ b1,
                                                   bf16* __restrict__ h1) {
    int wg = blockIdx.x * 8 + (threadIdx.x >> 5);      // b*G1 + g
    int lane = threadIdx.x & 31;
    int b = wg >> 10;
    const float* P = coords + (size_t)b * Np * 3;
    const float* C = centers1 + (size_t)wg * 3;
    float cx = C[0], cy = C[1], cz = C[2];

    int p = select_knn<Np, K1n>(P, cx, cy, cz, lane);

    const float* Q = P + (size_t)p * 3;
    float rx = Q[0] - cx, ry = Q[1] - cy, rz = Q[2] - cz;
    const float* F = feats + ((size_t)b * Np + p) * 3;
    float f0 = F[0], f1 = F[1], f2 = F[2];
    bf16* o = h1 + ((size_t)wg * K1n + lane) * 64;
#pragma unroll
    for (int n = 0; n < 64; ++n) {
        float acc = b1[n] + rx * w1[n] + ry * w1[64 + n] + rz * w1[128 + n]
                  + f0 * w1[192 + n] + f1 * w1[256 + n] + f2 * w1[320 + n];
        o[n] = (bf16)fmaxf(acc, 0.0f);
    }
}

// kNN-2 + gather + zero-pad to K=160: f2p bf16 (B*G2*K2, 160)
__global__ void __launch_bounds__(256) knn2_kernel(const float* __restrict__ centers1,
                                                   const float* __restrict__ x1,
                                                   bf16* __restrict__ f2p) {
    int wg = blockIdx.x * 8 + (threadIdx.x >> 5);      // b*G2 + g
    int lane = threadIdx.x & 31;
    int b = wg >> 8, g = wg & 255;
    const float* P = centers1 + (size_t)b * G1n * 3;
    const float* C = P + (size_t)g * 3;                 // centers2 = centers1[:, :256]
    float cx = C[0], cy = C[1], cz = C[2];

    int p = select_knn<G1n, K2n>(P, cx, cy, cz, lane);

    if (lane < K2n) {
        const float* Q = P + (size_t)p * 3;
        const float* F = x1 + ((size_t)b * G1n + p) * 128;
        bf16* o = f2p + ((size_t)wg * K2n + lane) * 160;
        o[0] = (bf16)(Q[0] - cx); o[1] = (bf16)(Q[1] - cy); o[2] = (bf16)(Q[2] - cz);
#pragma unroll
        for (int c = 0; c < 128; ++c) o[3 + c] = (bf16)F[c];
#pragma unroll
        for (int c = 131; c < 160; ++c) o[c] = (bf16)0.0f;
    }
}

// ---------------------------------------------------------------------------
// Generic WMMA bf16 GEMM: Out = act(A[M,K] @ Wp + bias) with Wp pre-packed in
// fragment layout. Block = 4 waves: 32 M-rows x 64 N-cols; each wave owns one
// 16-col N-tile and both row-tiles. A tile staged in LDS (async-to-LDS when
// available); B fragments loaded directly from global (2x b128 per lane).
// Optional fused max-pool over POOL consecutive rows.
// ---------------------------------------------------------------------------
template <int K, int POOL, bool RELU, bool OUTBF>
__global__ void __launch_bounds__(128)
gemm_bf16(const bf16* __restrict__ A, const bf16* __restrict__ Wp,
          const float* __restrict__ bias, void* __restrict__ Out, int Nn) {
    static_assert(K % 32 == 0, "K must be a multiple of 32");
    constexpr int ASTR = K + 8;                       // pad to break LDS bank conflicts
    __attribute__((aligned(16))) __shared__ bf16 As[32 * ASTR];

    const int tid = threadIdx.x;
    const int wave = tid >> 5, lane = tid & 31;
    const int half = lane >> 4, mr = lane & 15;
    const int mtile = blockIdx.x;                     // 32 rows
    const int nt = blockIdx.y * 4 + wave;             // this wave's 16-col N-tile

    // stream 32xK A tile into LDS (rows contiguous in global)
    const bf16* Ag = A + (size_t)mtile * 32 * K;
    __builtin_prefetch(Ag + 32 * K, 0, 1);            // next M-tile -> global_prefetch_b8
    for (int i = tid; i < 32 * (K / 8); i += 128) {
        int row = i / (K / 8), seg = i % (K / 8);
#ifdef HAVE_ASYNC_LDS
        __builtin_amdgcn_global_load_async_to_lds_b128(
            (__attribute__((address_space(1))) i4vec*)(Ag + (size_t)row * K + seg * 8),
            (__attribute__((address_space(3))) i4vec*)&As[row * ASTR + seg * 8], 0, 0);
#else
        *(uint4*)&As[row * ASTR + seg * 8] = *(const uint4*)&Ag[(size_t)row * K + seg * 8];
#endif
    }
#ifdef HAVE_ASYNC_LDS
#if __has_builtin(__builtin_amdgcn_s_wait_asynccnt)
    __builtin_amdgcn_s_wait_asynccnt(0);
#else
    asm volatile("s_wait_asynccnt 0x0" ::: "memory");
#endif
#endif
    __syncthreads();

    v8f c0 = {0.f,0.f,0.f,0.f,0.f,0.f,0.f,0.f};
    v8f c1 = {0.f,0.f,0.f,0.f,0.f,0.f,0.f,0.f};
    // per-lane base of this wave's packed B fragments (32 B per lane per k-tile)
    const uint4* Wl = (const uint4*)(Wp + (((size_t)nt * (K / 32)) * 32 + lane) * 16);

#pragma unroll
    for (int kk = 0; kk < K / 32; ++kk) {
        union { v16bf v; unsigned u[8]; } a0, a1;
        union { v16bf v; uint4 q[2]; } bb;
        bb.q[0] = Wl[kk * 64 + 0];                    // 32 lanes * 2 uint4 per ktile
        bb.q[1] = Wl[kk * 64 + 1];
#pragma unroll
        for (int j = 0; j < 8; ++j) {
            int kb = kk * 32 + ((j >> 2) << 4) + (half << 3) + ((j & 3) << 1);
            a0.u[j] = *(const unsigned*)&As[mr * ASTR + kb];
            a1.u[j] = *(const unsigned*)&As[(16 + mr) * ASTR + kb];
        }
        c0 = __builtin_amdgcn_wmma_f32_16x16x32_bf16(false, a0.v, false, bb.v,
                                                     (short)0, c0, false, false);
        c1 = __builtin_amdgcn_wmma_f32_16x16x32_bf16(false, a1.v, false, bb.v,
                                                     (short)0, c1, false, false);
    }

    const int colg = (nt << 4) + mr;
    const float bv = bias[colg];

    if (POOL == 1) {
#pragma unroll
        for (int r = 0; r < 8; ++r) {
            float v0 = c0[r] + bv, v1 = c1[r] + bv;
            if (RELU) { v0 = fmaxf(v0, 0.f); v1 = fmaxf(v1, 0.f); }
            size_t row0 = (size_t)mtile * 32 + r + 8 * half;
            size_t row1 = row0 + 16;
            if (OUTBF) {
                ((bf16*)Out)[row0 * Nn + colg] = (bf16)v0;
                ((bf16*)Out)[row1 * Nn + colg] = (bf16)v1;
            } else {
                ((float*)Out)[row0 * Nn + colg] = v0;
                ((float*)Out)[row1 * Nn + colg] = v1;
            }
        }
    } else {
        float m0 = -3.0e38f, m1 = -3.0e38f;
#pragma unroll
        for (int r = 0; r < 8; ++r) {
            float v0 = c0[r] + bv, v1 = c1[r] + bv;
            if (RELU) { v0 = fmaxf(v0, 0.f); v1 = fmaxf(v1, 0.f); }
            m0 = fmaxf(m0, v0); m1 = fmaxf(m1, v1);
        }
        if (POOL == 32) {                              // one group per 32-row tile
            float m = fmaxf(m0, m1);
            m = fmaxf(m, __shfl_xor(m, 16, 32));
            if (half == 0) {
                if (OUTBF) ((bf16*)Out)[(size_t)mtile * Nn + colg] = (bf16)m;
                else       ((float*)Out)[(size_t)mtile * Nn + colg] = m;
            }
        } else {                                        // POOL == 16: one group per row-tile
            m0 = fmaxf(m0, __shfl_xor(m0, 16, 32));
            m1 = fmaxf(m1, __shfl_xor(m1, 16, 32));
            if (half == 0) {
                if (OUTBF) {
                    ((bf16*)Out)[(size_t)(mtile * 2 + 0) * Nn + colg] = (bf16)m0;
                    ((bf16*)Out)[(size_t)(mtile * 2 + 1) * Nn + colg] = (bf16)m1;
                } else {
                    ((float*)Out)[(size_t)(mtile * 2 + 0) * Nn + colg] = m0;
                    ((float*)Out)[(size_t)(mtile * 2 + 1) * Nn + colg] = m1;
                }
            }
        }
    }
}

// ---------------------------------------------------------------------------
extern "C" void kernel_launch(void* const* d_in, const int* in_sizes, int n_in,
                              void* d_out, int out_size, void* d_ws, size_t ws_size,
                              hipStream_t stream) {
    const float* coords  = (const float*)d_in[0];
    const float* feats   = (const float*)d_in[1];
    const float* e1_w1   = (const float*)d_in[2];
    const float* e1_b1   = (const float*)d_in[3];
    const float* e1_w2   = (const float*)d_in[4];
    const float* e1_b2   = (const float*)d_in[5];
    const float* e1_w3   = (const float*)d_in[6];
    const float* e1_b3   = (const float*)d_in[7];
    const float* e2_w1   = (const float*)d_in[8];
    const float* e2_b1   = (const float*)d_in[9];
    const float* e2_w2   = (const float*)d_in[10];
    const float* e2_b2   = (const float*)d_in[11];
    const float* e2_w3   = (const float*)d_in[12];
    const float* e2_b3   = (const float*)d_in[13];

    char* wsp = (char*)d_ws;
    auto alloc = [&](size_t bytes) -> void* {
        void* p = (void*)wsp;
        wsp += (bytes + 255) & ~(size_t)255;
        return p;
    };
    float* centers1 = (float*)alloc((size_t)Bn * G1n * 3 * 4);
    bf16*  h1       = (bf16*) alloc((size_t)Bn * G1n * K1n * 64 * 2);   // 131072 x 64
    bf16*  h2p      = (bf16*) alloc((size_t)Bn * G1n * 128 * 2);        // 4096 x 128
    float* x1       = (float*)alloc((size_t)Bn * G1n * 128 * 4);        // 4096 x 128
    bf16*  f2p      = (bf16*) alloc((size_t)Bn * G2n * K2n * 160 * 2);  // 16384 x 160
    bf16*  e2h1     = (bf16*) alloc((size_t)Bn * G2n * K2n * 128 * 2);  // 16384 x 128
    bf16*  e2p      = (bf16*) alloc((size_t)Bn * G2n * 256 * 2);        // 1024 x 256
    bf16*  w2b      = (bf16*) alloc((size_t)64 * 128 * 2);
    bf16*  e1w3b    = (bf16*) alloc((size_t)128 * 128 * 2);
    bf16*  e2w1pb   = (bf16*) alloc((size_t)160 * 128 * 2);
    bf16*  e2w2b    = (bf16*) alloc((size_t)128 * 256 * 2);
    bf16*  e2w3b    = (bf16*) alloc((size_t)256 * 768 * 2);

    // Weight packing into WMMA fragment layout (independent of FPS)
    pack_w_kernel<<<(64 * 128 + 255) / 256, 256, 0, stream>>>(e1_w2, w2b, 64, 64, 128);
    pack_w_kernel<<<(128 * 128 + 255) / 256, 256, 0, stream>>>(e1_w3, e1w3b, 128, 128, 128);
    pack_w_kernel<<<(160 * 128 + 255) / 256, 256, 0, stream>>>(e2_w1, e2w1pb, 160, 131, 128);
    pack_w_kernel<<<(128 * 256 + 255) / 256, 256, 0, stream>>>(e2_w2, e2w2b, 128, 128, 256);
    pack_w_kernel<<<(256 * 768 + 255) / 256, 256, 0, stream>>>(e2_w3, e2w3b, 256, 256, 768);

    // Stage 1
    fps_kernel<<<Bn, 1024, 0, stream>>>(coords, centers1);
    knn1_kernel<<<(Bn * G1n) / 8, 256, 0, stream>>>(coords, feats, centers1, e1_w1, e1_b1, h1);
    // e1 layer2 (64->128, relu) + fused maxpool over K1=32 rows
    gemm_bf16<64, 32, true, true><<<dim3(4096, 2), 128, 0, stream>>>(h1, w2b, e1_b2, h2p, 128);
    // e1 layer3 (128->128, linear) -> x1 f32
    gemm_bf16<128, 1, false, false><<<dim3(128, 2), 128, 0, stream>>>(h2p, e1w3b, e1_b3, x1, 128);

    // Stage 2
    knn2_kernel<<<(Bn * G2n) / 8, 256, 0, stream>>>(centers1, x1, f2p);
    // e2 layer1 (160(pad of 131)->128, relu)
    gemm_bf16<160, 1, true, true><<<dim3(512, 2), 128, 0, stream>>>(f2p, e2w1pb, e2_b1, e2h1, 128);
    // e2 layer2 (128->256, relu) + fused maxpool over K2=16 rows
    gemm_bf16<128, 16, true, true><<<dim3(512, 4), 128, 0, stream>>>(e2h1, e2w2b, e2_b2, e2p, 256);
    // e2 layer3 (256->768, linear) -> d_out f32 (4,256,768)
    gemm_bf16<256, 1, false, false><<<dim3(32, 12), 128, 0, stream>>>(e2p, e2w3b, e2_b3, d_out, 768);
}
